// GCN_64561948393725
// MI455X (gfx1250) — matrix operand order repaired and checked by reading, not compiled, not run on hardware
//
#include <hip/hip_runtime.h>
#include <hip/hip_bf16.h>

#define BB 8
#define NN 2048
#define FF 128
#define DD 128

typedef __attribute__((ext_vector_type(16))) __bf16       v16bf;
typedef __attribute__((ext_vector_type(8)))  float        v8f;
typedef __attribute__((ext_vector_type(4)))  float        v4f;
typedef __attribute__((ext_vector_type(8)))  unsigned int v8u;

// ---------------------------------------------------------------------------
// Pass 1: dinv[b*N+n] = rsqrt(sum_m adj[b][n][m]); one wave per row.
// Streams adj nontemporally (134 MB, don't pollute L2).
// ---------------------------------------------------------------------------
__global__ void __launch_bounds__(256) gcn_rowsum(const float* __restrict__ adj,
                                                  float* __restrict__ dinv) {
    const int wave = threadIdx.x >> 5;
    const int lane = threadIdx.x & 31;
    const int row  = blockIdx.x * 8 + wave;              // 0 .. B*N-1
    const float* p = adj + (size_t)row * NN + lane * 4;
    float s = 0.f;
    #pragma unroll
    for (int i = 0; i < 16; ++i) {                       // 32 lanes * 4 * 16 = 2048
        v4f v = __builtin_nontemporal_load((const v4f*)(p + i * 128));
        s += v.x + v.y + v.z + v.w;
    }
    #pragma unroll
    for (int m = 16; m >= 1; m >>= 1) s += __shfl_xor(s, m, 32);
    if (lane == 0) {
        float r = (s > 0.f) ? rsqrtf(s) : 0.f;           // matches isinf->0 guard
        dinv[row] = r;
    }
}

// ---------------------------------------------------------------------------
// Pass 2: sfeatsT[b][d][m] = bf16( dinv[b*N+m] * (inputs[b][m]·W[:,d] + bias[d]) )
// Stored transposed (d-major) so pass 3's WMMA B-operand is contiguous per lane.
// One block of 128 threads per (b,m) row; X row staged in LDS.
// ---------------------------------------------------------------------------
__global__ void __launch_bounds__(128) gcn_feats(const float* __restrict__ inp,
                                                 const float* __restrict__ Wk,
                                                 const float* __restrict__ bias,
                                                 const float* __restrict__ dinv,
                                                 __bf16* __restrict__ sfT) {
    __shared__ float srow[FF];
    const int bm = blockIdx.x;                           // b*N + m
    const int d  = threadIdx.x;                          // 0..127
    srow[d] = inp[(size_t)bm * FF + d];
    __syncthreads();
    float acc = bias[d];
    #pragma unroll 8
    for (int f = 0; f < FF; ++f) acc = fmaf(srow[f], Wk[f * DD + d], acc);
    const int b = bm >> 11;                              // / 2048
    const int m = bm & (NN - 1);
    const float v = dinv[bm] * acc;
    sfT[((size_t)b * DD + d) * NN + m] = (__bf16)v;
}

// ---------------------------------------------------------------------------
// Pass 3: out[b][n][d] = relu( dinv[n] * sum_m adj[b][n][m] * sfeatsT[b][d][m] )
// One wave per 16(n) x 128(d) output strip. K-loop step 32, 8 WMMAs per step.
// All 8 B-tile fragments are register-blocked so their 16 b128 loads stay in
// flight together and the WMMAs can retire with staggered loadcnt waits.
// ---------------------------------------------------------------------------
__global__ void __launch_bounds__(128, 1) gcn_spmm(const float* __restrict__ adj,
                                                   const __bf16* __restrict__ sfT,
                                                   const float* __restrict__ dinv,
                                                   float* __restrict__ out) {
    const int w     = blockIdx.x * 4 + (threadIdx.x >> 5); // global wave id
    const int b     = w >> 7;                              // / 128 strips per batch
    const int strip = w & 127;
    const int n0    = strip * 16;
    const int lane  = threadIdx.x & 31;
    const int rl    = lane & 15;                           // row (A) / col (B,C)
    const int half  = lane >> 4;                           // K-half selector

    const v8f z = {0.f, 0.f, 0.f, 0.f, 0.f, 0.f, 0.f, 0.f};
    v8f acc[8] = {z, z, z, z, z, z, z, z};

    // A: lane rl holds row n0+rl; half=0 -> K 0-15, half=1 -> K 16-31
    const float*  aBase = adj + ((size_t)b * NN + n0 + rl) * NN + half * 16;
    // B: lane rl is column d%16; half selects K 0-15 vs 16-31; tile t adds 16 columns
    const __bf16* bBase = sfT + ((size_t)b * DD + rl) * NN + half * 16;

    for (int k0 = 0; k0 < NN; k0 += 32) {
        // --- issue A loads first (streaming, from HBM) ---
        v4f a0 = __builtin_nontemporal_load((const v4f*)(aBase + k0));
        v4f a1 = __builtin_nontemporal_load((const v4f*)(aBase + k0 + 4));
        v4f a2 = __builtin_nontemporal_load((const v4f*)(aBase + k0 + 8));
        v4f a3 = __builtin_nontemporal_load((const v4f*)(aBase + k0 + 12));
        if (k0 + 32 < NN) __builtin_prefetch(aBase + k0 + 32, 0, 3);

        // --- issue all 8 B-tile fragment loads (L2-resident) ---
        v8u braw[8];
        #pragma unroll
        for (int t = 0; t < 8; ++t)
            braw[t] = *(const v8u*)(bBase + (size_t)t * 16 * NN + k0);

        // --- build A fragment (contiguous K order, matches B) ---
        v16bf a;
        a[0]  = (__bf16)a0.x; a[1]  = (__bf16)a0.y; a[2]  = (__bf16)a0.z; a[3]  = (__bf16)a0.w;
        a[4]  = (__bf16)a1.x; a[5]  = (__bf16)a1.y; a[6]  = (__bf16)a1.z; a[7]  = (__bf16)a1.w;
        a[8]  = (__bf16)a2.x; a[9]  = (__bf16)a2.y; a[10] = (__bf16)a2.z; a[11] = (__bf16)a2.w;
        a[12] = (__bf16)a3.x; a[13] = (__bf16)a3.y; a[14] = (__bf16)a3.z; a[15] = (__bf16)a3.w;

        // --- 8 WMMAs, one per 16-column D tile ---
        #pragma unroll
        for (int t = 0; t < 8; ++t) {
            v16bf bf = __builtin_bit_cast(v16bf, braw[t]);
            acc[t] = __builtin_amdgcn_wmma_f32_16x16x32_bf16(
                false, a, false, bf, (short)0, acc[t], false, false);
        }
    }

    // Epilogue: C layout -> row = 8*half + j, col = rl (+16*t). Scale + relu.
    const float* dv = dinv + b * NN + n0 + half * 8;
    v4f d0 = *(const v4f*)dv;
    v4f d1 = *(const v4f*)(dv + 4);
    const float ds[8] = {d0.x, d0.y, d0.z, d0.w, d1.x, d1.y, d1.z, d1.w};
    float* ob = out + ((size_t)b * NN + n0 + half * 8) * DD + rl;
    #pragma unroll
    for (int t = 0; t < 8; ++t) {
        #pragma unroll
        for (int j = 0; j < 8; ++j) {
            float v = acc[t][j] * ds[j];
            ob[(size_t)j * DD + t * 16] = fmaxf(v, 0.f);
        }
    }
}

// ---------------------------------------------------------------------------
extern "C" void kernel_launch(void* const* d_in, const int* in_sizes, int n_in,
                              void* d_out, int out_size, void* d_ws, size_t ws_size,
                              hipStream_t stream) {
    const float* inputs = (const float*)d_in[0];   // [B,N,F]
    const float* adj    = (const float*)d_in[1];   // [B,N,N]
    const float* Wk     = (const float*)d_in[2];   // [F,D]
    const float* bias   = (const float*)d_in[3];   // [D]
    float* out = (float*)d_out;                    // [B,N,D] fp32

    // ws layout: [0, 4MB) sfeatsT bf16, then dinv fp32 (64 KB)
    __bf16* sfT  = (__bf16*)d_ws;
    float*  dinv = (float*)((char*)d_ws + (size_t)BB * DD * NN * sizeof(__bf16));

    gcn_rowsum<<<BB * NN / 8, 256, 0, stream>>>(adj, dinv);
    gcn_feats <<<BB * NN,     128, 0, stream>>>(inputs, Wk, bias, dinv, sfT);
    gcn_spmm  <<<BB * NN / 64, 128, 0, stream>>>(adj, sfT, dinv, out);
}
